// SequentialModeling_56264071578180
// MI455X (gfx1250) — matrix operand order repaired and checked by reading, not compiled, not run on hardware
//
#include <hip/hip_runtime.h>
#include <hip/hip_bf16.h>

// ---------------------------------------------------------------------------
// Mamba block forward for MI455X (gfx1250), wave32 + WMMA f16->f32,
// async global->LDS tile fills (ASYNCcnt), double-buffered LDS, BK=64.
// Shapes: B=4, L=2048, D_MODEL=1024, D_INNER=2048, D_STATE=16, D_CONV=4.
// ---------------------------------------------------------------------------

#define BATCH    4
#define SEQLEN   2048
#define DMODEL   1024
#define DINNER   2048
#define DSTATE   16
#define MROWS    (BATCH * SEQLEN)      // 8192
#define NPAD_XPROJ 128                 // x_proj output padded to one 128-tile

#define BK       64                    // K per LDS stage (2 WMMA sub-steps)
#define LDSW     72                    // halves per LDS row: 64 data + 8 pad
                                       // (144B stride -> 36-dword bank stride,
                                       //  conflict-free for 16-lane groups)
#define TILE_B   (128 * LDSW * 2)      // bytes per LDS tile buffer (18432)

typedef __attribute__((ext_vector_type(16))) _Float16 v16h;
typedef __attribute__((ext_vector_type(8)))  float    v8f;

// CDNA5 async global->LDS load, 16B per lane, tracked by ASYNCcnt
__device__ __forceinline__ void async_ld_b128(uint32_t lds_off, const void* g) {
    asm volatile("global_load_async_to_lds_b128 %0, %1, off"
                 :: "v"(lds_off), "v"(g) : "memory");
}
__device__ __forceinline__ void wait_async0() {
    asm volatile("s_wait_asynccnt 0x0" ::: "memory");
}

// ---------------------------------------------------------------------------
// 1) LayerNorm (fp32 in) -> f16 out, one block per row of 1024
// ---------------------------------------------------------------------------
__global__ __launch_bounds__(256) void ln_kernel(const float* __restrict__ x,
                                                 const float* __restrict__ g,
                                                 const float* __restrict__ b,
                                                 _Float16* __restrict__ xn) {
    __shared__ float s1[256], s2[256];
    const int row = blockIdx.x;
    const float* xr = x + (size_t)row * DMODEL;
    float a = 0.f, q = 0.f;
    for (int i = threadIdx.x; i < DMODEL; i += 256) {
        float v = xr[i];
        a += v; q += v * v;
    }
    s1[threadIdx.x] = a; s2[threadIdx.x] = q;
    __syncthreads();
    for (int st = 128; st > 0; st >>= 1) {
        if (threadIdx.x < st) {
            s1[threadIdx.x] += s1[threadIdx.x + st];
            s2[threadIdx.x] += s2[threadIdx.x + st];
        }
        __syncthreads();
    }
    const float mu  = s1[0] * (1.f / DMODEL);
    const float var = s2[0] * (1.f / DMODEL) - mu * mu;
    const float rs  = rsqrtf(var + 1e-5f);
    _Float16* xo = xn + (size_t)row * DMODEL;
    for (int i = threadIdx.x; i < DMODEL; i += 256)
        xo[i] = (_Float16)((xr[i] - mu) * rs * g[i] + b[i]);
}

// ---------------------------------------------------------------------------
// 2) Tiled weight convert+transpose: W (K x N fp32) -> Wt (Npad x K f16).
//    Coalesced both directions via a 32x32 LDS tile; n >= N rows zero-filled.
//    Launch: block (32,8), grid (K/32, Npad/32).
// ---------------------------------------------------------------------------
__global__ __launch_bounds__(256) void transpose_cvt_kernel(
    const float* __restrict__ W, _Float16* __restrict__ Wt,
    int K, int N) {
    __shared__ float tile[32][33];
    const int k0 = blockIdx.x * 32, n0 = blockIdx.y * 32;
    const int tx = threadIdx.x, ty = threadIdx.y;       // 32 x 8
#pragma unroll
    for (int j = 0; j < 32; j += 8) {
        const int k = k0 + ty + j, n = n0 + tx;
        tile[ty + j][tx] = (n < N) ? W[(size_t)k * N + n] : 0.f;
    }
    __syncthreads();
#pragma unroll
    for (int j = 0; j < 32; j += 8) {
        const int n = n0 + ty + j, k = k0 + tx;
        Wt[(size_t)n * K + k] = (_Float16)tile[tx][ty + j];
    }
}

// ---------------------------------------------------------------------------
// 3) WMMA GEMM: C[m,n] = sum_k A[m,k] * Bt[n,k]  (+ optional addend)
//    A: (M x K) f16 row-major, Bt: (N x K) f16 row-major, C: (M x N) f32.
//    256 threads = 8 waves; tile 128x128; BK=64; double-buffered LDS filled
//    with global_load_async_to_lds_b128; wave = 2x4 grid of 16x16 WMMA tiles,
//    16 WMMAs per stage between barriers.
// ---------------------------------------------------------------------------
__global__ __launch_bounds__(256) void gemm_wmma_kernel(
    const _Float16* __restrict__ A, const _Float16* __restrict__ Bt,
    float* __restrict__ C, const float* __restrict__ addend,
    int M, int N, int K) {
    __shared__ _Float16 sA[2][128][LDSW];
    __shared__ _Float16 sB[2][128][LDSW];

    const int tid  = threadIdx.x;
    const int lane = tid & 31;
    const int wave = tid >> 5;
    const int m0 = blockIdx.y * 128;
    const int n0 = blockIdx.x * 128;
    const int wm = (wave & 3) * 32;   // wave row offset within tile
    const int wn = (wave >> 2) * 64;  // wave col offset within tile

    // cooperative fill: tile = 128 rows x 8 chunks(16B); thread owns chunk
    // (fRow + rr*32, fC8) for rr = 0..3
    const int fRow = tid >> 3;        // 0..31
    const int fC8  = tid & 7;         // 16B chunk within 128B data row

    // LDS byte offsets (generic shared pointer low 32 bits == LDS address)
    const uint32_t sA0 = (uint32_t)(uintptr_t)&sA[0][0][0];
    const uint32_t sB0 = (uint32_t)(uintptr_t)&sB[0][0][0];
    const uint32_t laA = sA0 + (uint32_t)(fRow * (LDSW * 2) + fC8 * 16);
    const uint32_t laB = sB0 + (uint32_t)(fRow * (LDSW * 2) + fC8 * 16);

    v8f acc[2][4];
    const v8f vzero = {0.f, 0.f, 0.f, 0.f, 0.f, 0.f, 0.f, 0.f};
#pragma unroll
    for (int r = 0; r < 2; ++r)
#pragma unroll
        for (int c = 0; c < 4; ++c) acc[r][c] = vzero;

    const int frow = lane & 15;            // M (A) / N (B) within 16
    const int kHi  = lane >> 4;            // lane-half selector

    auto fillTile = [&](int buf, int kt) {
        const uint32_t bo = (uint32_t)buf * TILE_B;
#pragma unroll
        for (int rr = 0; rr < 4; ++rr) {
            const uint32_t lo = bo + (uint32_t)(rr * 32 * (LDSW * 2));
            async_ld_b128(laA + lo,
                A  + (size_t)(m0 + fRow + rr * 32) * K + kt + fC8 * 8);
            async_ld_b128(laB + lo,
                Bt + (size_t)(n0 + fRow + rr * 32) * K + kt + fC8 * 8);
        }
    };

    const int nIter = K >> 6;
    fillTile(0, 0);

    for (int it = 0; it < nIter; ++it) {
        wait_async0();          // this wave's fills for buf are in LDS
        __syncthreads();        // every wave's fills are in LDS
        if (it + 1 < nIter) fillTile((it + 1) & 1, (it + 1) << 6);
        const int buf = it & 1;

#pragma unroll
        for (int s = 0; s < 2; ++s) {        // two 32-K WMMA sub-steps
            const int kA = s * 32 + kHi * 8;   // A chunk base (halves)
            const int kB = s * 32 + kHi * 8;   // B chunk base (halves)

            // A fragments (ISA 7.12.2): two 16B chunks per lane
            v16h af[2];
#pragma unroll
            for (int r = 0; r < 2; ++r) {
                const _Float16* p = &sA[buf][wm + r * 16 + frow][0];
                ((uint4*)&af[r])[0] = *(const uint4*)(p + kA);
                ((uint4*)&af[r])[1] = *(const uint4*)(p + kA + 16);
            }
            // B fragments: two 16B chunks per lane (same K order as A)
            v16h bf[4];
#pragma unroll
            for (int c = 0; c < 4; ++c) {
                const _Float16* p = &sB[buf][wn + c * 16 + frow][0];
                ((uint4*)&bf[c])[0] = *(const uint4*)(p + kB);
                ((uint4*)&bf[c])[1] = *(const uint4*)(p + kB + 16);
            }
#pragma unroll
            for (int c = 0; c < 4; ++c)
#pragma unroll
                for (int r = 0; r < 2; ++r)
                    acc[r][c] = __builtin_amdgcn_wmma_f32_16x16x32_f16(
                        false, af[r], false, bf[c], (short)0, acc[r][c],
                        false, false);
        }
    }

    // C layout: lane<16 -> rows 0..7 (vgpr i = M=i), lane>=16 -> rows 8..15
    const int cRow = kHi * 8;
    const int cCol = frow;
    if (addend) {
#pragma unroll
        for (int r = 0; r < 2; ++r)
#pragma unroll
            for (int c = 0; c < 4; ++c) {
                const size_t mbase = (size_t)(m0 + wm + r * 16 + cRow);
                const int nn = n0 + wn + c * 16 + cCol;
#pragma unroll
                for (int i = 0; i < 8; ++i) {
                    size_t idx = (mbase + i) * (size_t)N + nn;
                    C[idx] = acc[r][c][i] + addend[idx];
                }
            }
    } else {
#pragma unroll
        for (int r = 0; r < 2; ++r)
#pragma unroll
            for (int c = 0; c < 4; ++c) {
                const size_t mbase = (size_t)(m0 + wm + r * 16 + cRow);
                const int nn = n0 + wn + c * 16 + cCol;
#pragma unroll
                for (int i = 0; i < 8; ++i)
                    C[(mbase + i) * (size_t)N + nn] = acc[r][c][i];
            }
    }
}

// ---------------------------------------------------------------------------
// 4) Causal depthwise conv (k=4) + bias + SiLU. Reads u = xz[:, 0:2048].
// ---------------------------------------------------------------------------
__global__ void conv_silu_kernel(const float* __restrict__ xz,
                                 const float* __restrict__ cw,
                                 const float* __restrict__ cb,
                                 float* __restrict__ u_act,
                                 _Float16* __restrict__ u_h) {
    size_t i = (size_t)blockIdx.x * blockDim.x + threadIdx.x;
    if (i >= (size_t)MROWS * DINNER) return;
    const int    c    = (int)(i & (DINNER - 1));
    const size_t grow = i >> 11;                 // global row = b*L + l
    const int    l    = (int)(grow & (SEQLEN - 1));
    const size_t bb   = grow - l;                // b*L
    float acc = cb[c];
#pragma unroll
    for (int j = 0; j < 4; ++j) {
        int lj = l - 3 + j;
        if (lj >= 0)
            acc += cw[c * 4 + j] * xz[(bb + lj) * (size_t)(2 * DINNER) + c];
    }
    float s = acc * (1.f / (1.f + __expf(-acc)));   // SiLU
    u_act[i] = s;
    u_h[i]   = (_Float16)s;
}

// ---------------------------------------------------------------------------
// 5) delta = softplus(dt_raw + bias)   (in place)
// ---------------------------------------------------------------------------
__global__ void softplus_bias_kernel(float* __restrict__ dt,
                                     const float* __restrict__ bias,
                                     size_t total) {
    size_t i = (size_t)blockIdx.x * blockDim.x + threadIdx.x;
    if (i >= total) return;
    const int c = (int)(i & (DINNER - 1));
    float v = dt[i] + bias[c];
    dt[i] = (v > 20.f) ? v : log1pf(__expf(v));
}

// ---------------------------------------------------------------------------
// 6) Selective scan: one thread per (batch, channel); h[16] in registers.
//    Software-pipelined t+1 loads + global_prefetch 16 rows ahead to hide
//    memory latency in the serial recurrence.
// ---------------------------------------------------------------------------
__global__ __launch_bounds__(256) void scan_kernel(
    const float* __restrict__ u, const float* __restrict__ dt,
    const float* __restrict__ bm_pad, const float* __restrict__ A_log,
    const float* __restrict__ Dp, float* __restrict__ y) {
    const int idx = blockIdx.x * blockDim.x + threadIdx.x;   // 0..8191
    const int b = idx >> 11;
    const int c = idx & (DINNER - 1);
    float Arow[DSTATE];
#pragma unroll
    for (int s = 0; s < DSTATE; ++s)
        Arow[s] = -__expf(A_log[c * DSTATE + s]);
    const float Dc = Dp[c];
    float h[DSTATE];
#pragma unroll
    for (int s = 0; s < DSTATE; ++s) h[s] = 0.f;

    const size_t rbase = (size_t)b * SEQLEN;
    const float* up = u      + rbase * DINNER + c;
    const float* dp = dt     + rbase * DINNER + c;
    const float* bp = bm_pad + rbase * NPAD_XPROJ + DSTATE;
    float*       yp = y      + rbase * DINNER + c;

    // pipeline prologue: t = 0 operands
    float uv = up[0];
    float dv = dp[0];
    float4 b0 = ((const float4*)bp)[0];
    float4 b1 = ((const float4*)bp)[1];
    float4 b2 = ((const float4*)bp)[2];
    float4 b3 = ((const float4*)bp)[3];

    for (int t = 0; t < SEQLEN; ++t) {
        // issue next-step loads before the recurrence (depth-1 pipeline)
        const int tn = (t + 1 < SEQLEN) ? t + 1 : t;
        const float uv_n = up[(size_t)tn * DINNER];
        const float dv_n = dp[(size_t)tn * DINNER];
        const float4* bpn = (const float4*)(bp + (size_t)tn * NPAD_XPROJ);
        const float4 nb0 = bpn[0], nb1 = bpn[1], nb2 = bpn[2], nb3 = bpn[3];

        // long-range prefetch into caches (global_prefetch_b8, no LOADcnt)
        if (t + 16 < SEQLEN) {
            __builtin_prefetch(up + (size_t)(t + 16) * DINNER, 0, 0);
            __builtin_prefetch(dp + (size_t)(t + 16) * DINNER, 0, 0);
            __builtin_prefetch(bp + (size_t)(t + 16) * NPAD_XPROJ, 0, 0);
        }

        const float Bv[DSTATE] = {b0.x, b0.y, b0.z, b0.w, b1.x, b1.y, b1.z, b1.w,
                                  b2.x, b2.y, b2.z, b2.w, b3.x, b3.y, b3.z, b3.w};
        const float ud = uv * dv;
        float accv = 0.f;
#pragma unroll
        for (int s = 0; s < DSTATE; ++s) {
            h[s] = h[s] * __expf(dv * Arow[s]) + ud * Bv[s];
            accv += h[s];
        }
        yp[(size_t)t * DINNER] = accv + uv * Dc;

        uv = uv_n; dv = dv_n;
        b0 = nb0; b1 = nb1; b2 = nb2; b3 = nb3;
    }
}

// ---------------------------------------------------------------------------
// 7) gate: yg = y * silu(z),  z = xz[:, 2048:4096]; f16 out for final GEMM
// ---------------------------------------------------------------------------
__global__ void gate_kernel(const float* __restrict__ y,
                            const float* __restrict__ xz,
                            _Float16* __restrict__ yg) {
    size_t i = (size_t)blockIdx.x * blockDim.x + threadIdx.x;
    if (i >= (size_t)MROWS * DINNER) return;
    const size_t row = i >> 11;
    const int    c   = (int)(i & (DINNER - 1));
    float zv = xz[row * (size_t)(2 * DINNER) + DINNER + c];
    float gz = zv * (1.f / (1.f + __expf(-zv)));
    yg[i] = (_Float16)(y[i] * gz);
}

// ---------------------------------------------------------------------------
// Launch
// ---------------------------------------------------------------------------
extern "C" void kernel_launch(void* const* d_in, const int* in_sizes, int n_in,
                              void* d_out, int out_size, void* d_ws, size_t ws_size,
                              hipStream_t stream) {
    (void)in_sizes; (void)n_in; (void)out_size; (void)ws_size;
    const float* x         = (const float*)d_in[0];
    const float* ln_gamma  = (const float*)d_in[1];
    const float* ln_beta   = (const float*)d_in[2];
    const float* in_proj_w = (const float*)d_in[3];
    const float* conv_w    = (const float*)d_in[4];
    const float* conv_b    = (const float*)d_in[5];
    const float* x_proj_w  = (const float*)d_in[6];
    const float* dt_proj_w = (const float*)d_in[7];
    const float* dt_proj_b = (const float*)d_in[8];
    const float* A_log     = (const float*)d_in[9];
    const float* D_param   = (const float*)d_in[10];
    const float* out_proj_w= (const float*)d_in[11];
    float* out = (float*)d_out;

    // workspace carve-out (256B aligned)
    char* ws = (char*)d_ws;
    size_t off = 0;
    auto take = [&](size_t bytes) -> void* {
        void* p = ws + off;
        off = (off + bytes + 255) & ~(size_t)255;
        return p;
    };
    _Float16* xn       = (_Float16*)take((size_t)MROWS * DMODEL * 2);
    _Float16* inprojT  = (_Float16*)take((size_t)(2 * DINNER) * DMODEL * 2);
    _Float16* dtprojT  = (_Float16*)take((size_t)DINNER * DINNER * 2);
    _Float16* outprojT = (_Float16*)take((size_t)DMODEL * DINNER * 2);
    _Float16* xprojT   = (_Float16*)take((size_t)NPAD_XPROJ * DINNER * 2);
    float*    xz       = (float*)   take((size_t)MROWS * 2 * DINNER * 4);
    float*    u_act    = (float*)   take((size_t)MROWS * DINNER * 4);
    _Float16* u_h      = (_Float16*)take((size_t)MROWS * DINNER * 2);
    float*    dt       = (float*)   take((size_t)MROWS * DINNER * 4);
    float*    bm_pad   = (float*)   take((size_t)MROWS * NPAD_XPROJ * 4);
    float*    yb       = (float*)   take((size_t)MROWS * DINNER * 4);
    _Float16* yg       = (_Float16*)take((size_t)MROWS * DINNER * 2);

    const size_t EW = (size_t)MROWS * DINNER;          // 16.7M elementwise items
    const int    EB = (int)((EW + 255) / 256);

    // 1) LayerNorm -> f16
    ln_kernel<<<MROWS, 256, 0, stream>>>(x, ln_gamma, ln_beta, xn);

    // 2) weights -> transposed f16 (32x32 LDS tiles, coalesced)
    {
        dim3 blk(32, 8);
        transpose_cvt_kernel<<<dim3(DMODEL / 32, (2 * DINNER) / 32), blk, 0, stream>>>(
            in_proj_w, inprojT, DMODEL, 2 * DINNER);
        transpose_cvt_kernel<<<dim3(DINNER / 32, DINNER / 32), blk, 0, stream>>>(
            dt_proj_w, dtprojT, DINNER, DINNER);
        transpose_cvt_kernel<<<dim3(DINNER / 32, DMODEL / 32), blk, 0, stream>>>(
            out_proj_w, outprojT, DINNER, DMODEL);
        transpose_cvt_kernel<<<dim3(DINNER / 32, NPAD_XPROJ / 32), blk, 0, stream>>>(
            x_proj_w, xprojT, DINNER, 2 * DSTATE);
    }

    // 3) xz = xn @ in_proj_w           (8192 x 4096 x 1024)
    gemm_wmma_kernel<<<dim3((2 * DINNER) / 128, MROWS / 128), 256, 0, stream>>>(
        xn, inprojT, xz, nullptr, MROWS, 2 * DINNER, DMODEL);

    // 4) depthwise causal conv + SiLU
    conv_silu_kernel<<<EB, 256, 0, stream>>>(xz, conv_w, conv_b, u_act, u_h);

    // 5) dt_raw = u @ dt_proj_w        (8192 x 2048 x 2048), then softplus+bias
    gemm_wmma_kernel<<<dim3(DINNER / 128, MROWS / 128), 256, 0, stream>>>(
        u_h, dtprojT, dt, nullptr, MROWS, DINNER, DINNER);
    softplus_bias_kernel<<<EB, 256, 0, stream>>>(dt, dt_proj_b, EW);

    // 6) bm_pad = u @ x_proj_w (padded) (8192 x 128 x 2048); Bm = cols 16..31
    gemm_wmma_kernel<<<dim3(NPAD_XPROJ / 128, MROWS / 128), 256, 0, stream>>>(
        u_h, xprojT, bm_pad, nullptr, MROWS, NPAD_XPROJ, DINNER);

    // 7) selective scan
    scan_kernel<<<(BATCH * DINNER) / 256, 256, 0, stream>>>(
        u_act, dt, bm_pad, A_log, D_param, yb);

    // 8) gate with silu(z)
    gate_kernel<<<EB, 256, 0, stream>>>(yb, xz, yg);

    // 9) out = yg @ out_proj_w + residual   (8192 x 1024 x 2048)
    gemm_wmma_kernel<<<dim3(DMODEL / 128, MROWS / 128), 256, 0, stream>>>(
        yg, outprojT, out, x, MROWS, DMODEL, DINNER);
}
